// ScaledDotProductAttention_45792941310419
// MI455X (gfx1250) — matrix operand order repaired and checked by reading, not compiled, not run on hardware
//
#include <hip/hip_runtime.h>

typedef __attribute__((ext_vector_type(16))) __bf16          v16bf;
typedef __attribute__((ext_vector_type(8)))  float           v8f;
typedef __attribute__((ext_vector_type(16))) unsigned short  v16u;

#define B_    8
#define L_    2048
#define D_    64
#define KT    64            // keys per staged tile
#define QW    16            // query rows per wave
#define WAVES 8
#define QB    (QW * WAVES)  // 128 query rows per block

// f32 -> bf16 through the native converter
__device__ __forceinline__ unsigned short f2bf(float f) {
  return __builtin_bit_cast(unsigned short, (__bf16)f);
}
__device__ __forceinline__ unsigned pk2bf(float lo, float hi) {
  return (unsigned)f2bf(lo) | ((unsigned)f2bf(hi) << 16);
}

__device__ __forceinline__ v16u lds_ld16(const unsigned short* p) {
  union { uint4 q[2]; v16u v; } u;
  u.q[0] = ((const uint4*)p)[0];
  u.q[1] = ((const uint4*)p)[1];
  return u.v;
}

__device__ __forceinline__ void pack8(v16u& v, int base, float4 a, float4 b, float sc) {
  v[base + 0] = f2bf(a.x * sc); v[base + 1] = f2bf(a.y * sc);
  v[base + 2] = f2bf(a.z * sc); v[base + 3] = f2bf(a.w * sc);
  v[base + 4] = f2bf(b.x * sc); v[base + 5] = f2bf(b.y * sc);
  v[base + 6] = f2bf(b.z * sc); v[base + 7] = f2bf(b.w * sc);
}

// DPP lane permute within rows of 16 (no LDS traffic)
template <int CTRL>
__device__ __forceinline__ float dppf(float x) {
  int r = __builtin_amdgcn_update_dpp(0, __builtin_bit_cast(int, x), CTRL, 0xf, 0xf, true);
  return __builtin_bit_cast(float, r);
}
// max-reduce across each 16-lane half; result replicated to all lanes of the half
__device__ __forceinline__ float rowmax16(float x) {
  x = fmaxf(x, dppf<0xB1>(x));   // quad_perm [1,0,3,2]  (xor1)
  x = fmaxf(x, dppf<0x4E>(x));   // quad_perm [2,3,0,1]  (xor2)
  x = fmaxf(x, dppf<0x141>(x));  // row_half_mirror      (xor4: quads uniform)
  x = fmaxf(x, dppf<0x140>(x));  // row_mirror           (xor8: octs uniform)
  return x;
}

__global__ __launch_bounds__(256) void
sdpa_flash_wmma_bf16(const float* __restrict__ Qp, const float* __restrict__ Kp,
                     const float* __restrict__ Vp, float* __restrict__ Op) {
  // double-buffered K tile [key][d] and transposed V tile [d][key], bf16,
  // stride 72 elements (144B, multiple of 16B for ds_load_b128)
  __shared__ __align__(16) unsigned short Ks[2][KT][72];
  __shared__ __align__(16) unsigned short Vt[2][D_][72];
  __shared__ __align__(16) unsigned short Ps[WAVES][QW][64];  // per-wave P staging

  const int tid  = threadIdx.x;
  const int wv   = tid >> 5;
  const int lane = tid & 31;
  const int hb   = lane >> 4;   // 16-lane half
  const int ln   = lane & 15;

  const int b  = blockIdx.x / (L_ / QB);
  const int q0 = (blockIdx.x % (L_ / QB)) * QB;
  const size_t bb = (size_t)b * L_ * D_;

  // staging coordinates: K by single rows, V by row pairs (packed b32 stores)
  const int srow = tid >> 4;          // 0..15 (+16 per i)
  const int sc4  = (tid & 15) * 4;    // column (floats)

  // ---- Q A-fragments, scale 1/sqrt(64)*log2(e) folded in ----
  const float kSc = 0.125f * 1.44269504088896f;
  const float* qrow = Qp + bb + (size_t)(q0 + wv * QW + ln) * D_;
  v16u qf[2];
#pragma unroll
  for (int c = 0; c < 2; ++c) {
    const int b0 = 32 * c + 16 * hb;  // elems 0..7  -> d = b0..b0+7
    const int b1 = b0 + 8;            // elems 8..15 -> d = b1..b1+7
    pack8(qf[c], 0, *(const float4*)(qrow + b0), *(const float4*)(qrow + b0 + 4), kSc);
    pack8(qf[c], 8, *(const float4*)(qrow + b1), *(const float4*)(qrow + b1 + 4), kSc);
  }

  // all-ones bf16 B-fragment: D = P x 1 puts rowsum(P) in every column (lane)
  v16u onef;
#pragma unroll
  for (int j = 0; j < 16; ++j) onef[j] = 0x3F80;

  float m_run[8];
  v8f oacc[4], lacc;
#pragma unroll
  for (int r = 0; r < 8; ++r) m_run[r] = -3.0e38f;
#pragma unroll
  for (int t = 0; t < 4; ++t) oacc[t] = (v8f){0.f,0.f,0.f,0.f,0.f,0.f,0.f,0.f};
  lacc = (v8f){0.f,0.f,0.f,0.f,0.f,0.f,0.f,0.f};

  // ---- software pipeline: prefetch tile 0 into registers ----
  float4 kr[4], vr[4];
#pragma unroll
  for (int i = 0; i < 4; ++i)
    kr[i] = *(const float4*)(Kp + bb + (size_t)(srow + 16 * i) * D_ + sc4);
#pragma unroll
  for (int j = 0; j < 2; ++j) {
    const int pi = tid + j * 256;                   // 0..511 row-pairs*cols
    const int pr = pi >> 4, pc4 = (pi & 15) * 4;
    vr[2*j]   = *(const float4*)(Vp + bb + (size_t)(2*pr)     * D_ + pc4);
    vr[2*j+1] = *(const float4*)(Vp + bb + (size_t)(2*pr + 1) * D_ + pc4);
  }

  for (int kt = 0, it = 0; kt < L_; kt += KT, ++it) {
    const int cur = it & 1;

    // ---- drain prefetch registers into LDS (bf16; V transposed) ----
#pragma unroll
    for (int i = 0; i < 4; ++i) {
      uint2 pk;
      pk.x = pk2bf(kr[i].x, kr[i].y);
      pk.y = pk2bf(kr[i].z, kr[i].w);
      *(uint2*)&Ks[cur][srow + 16 * i][sc4] = pk;
    }
#pragma unroll
    for (int j = 0; j < 2; ++j) {
      const int pi = tid + j * 256;
      const int pr = pi >> 4, pc4 = (pi & 15) * 4;
      union { float4 f; float a[4]; } lo, hi;
      lo.f = vr[2*j]; hi.f = vr[2*j+1];
#pragma unroll
      for (int k = 0; k < 4; ++k)
        *(unsigned*)&Vt[cur][pc4 + k][2 * pr] = pk2bf(lo.a[k], hi.a[k]);
    }
    // ---- issue global loads for the next tile (retire under compute) ----
    if (kt + KT < L_) {
#pragma unroll
      for (int i = 0; i < 4; ++i)
        kr[i] = *(const float4*)(Kp + bb + (size_t)(kt + KT + srow + 16 * i) * D_ + sc4);
#pragma unroll
      for (int j = 0; j < 2; ++j) {
        const int pi = tid + j * 256;
        const int pr = pi >> 4, pc4 = (pi & 15) * 4;
        vr[2*j]   = *(const float4*)(Vp + bb + (size_t)(kt + KT + 2*pr)     * D_ + pc4);
        vr[2*j+1] = *(const float4*)(Vp + bb + (size_t)(kt + KT + 2*pr + 1) * D_ + pc4);
      }
    }
    __syncthreads();  // stores of `cur` visible; prior readers of `cur` done

    // ---- S(log2 domain) = Qs K^T : c-outer, 4 independent accumulators ----
    v8f s[4];
#pragma unroll
    for (int sub = 0; sub < 4; ++sub) s[sub] = (v8f){0.f,0.f,0.f,0.f,0.f,0.f,0.f,0.f};
#pragma unroll
    for (int c = 0; c < 2; ++c) {
#pragma unroll
      for (int sub = 0; sub < 4; ++sub) {
        v16u kf = lds_ld16(&Ks[cur][sub * 16 + ln][32 * c + 16 * hb]);
        s[sub] = __builtin_amdgcn_wmma_f32_16x16x32_bf16(
            false, __builtin_bit_cast(v16bf, qf[c]),
            false, __builtin_bit_cast(v16bf, kf),
            (short)0, s[sub], false, false);
      }
    }

    // ---- online softmax (base-2): DPP max-reduce, no LDS ----
    float mn[8];
    v8f alv;
#pragma unroll
    for (int r = 0; r < 8; ++r) {
      float mx = fmaxf(fmaxf(s[0][r], s[1][r]), fmaxf(s[2][r], s[3][r]));
      mx = rowmax16(mx);
      mn[r] = fmaxf(m_run[r], mx);
      alv[r] = __builtin_amdgcn_exp2f(m_run[r] - mn[r]);
      m_run[r] = mn[r];
    }
#pragma unroll
    for (int sub = 0; sub < 4; ++sub) {
#pragma unroll
      for (int r = 0; r < 8; ++r) {
        float w = __builtin_amdgcn_exp2f(s[sub][r] - mn[r]);
        Ps[wv][r + 8 * hb][sub * 16 + ln] = f2bf(w);  // C-layout -> row-major
      }
    }
    lacc = lacc * alv;
#pragma unroll
    for (int t = 0; t < 4; ++t) oacc[t] = oacc[t] * alv;

    // per-wave LDS round trip: DS is in-order within a wave; drain before reload
    asm volatile("s_wait_dscnt 0" ::: "memory");

    // ---- O += P*V and l += P*1 : A = P (16x32 bf16) ----
#pragma unroll
    for (int c = 0; c < 2; ++c) {
      union { uint4 q[2]; v16u v; } pu;
      pu.q[0] = *(const uint4*)&Ps[wv][ln][32 * c + 16 * hb];
      pu.q[1] = *(const uint4*)&Ps[wv][ln][32 * c + 16 * hb + 8];
      v16bf pf = __builtin_bit_cast(v16bf, pu.v);
#pragma unroll
      for (int t = 0; t < 4; ++t) {
        v16u vf = lds_ld16(&Vt[cur][16 * t + ln][32 * c + 16 * hb]);
        oacc[t] = __builtin_amdgcn_wmma_f32_16x16x32_bf16(
            false, pf, false, __builtin_bit_cast(v16bf, vf),
            (short)0, oacc[t], false, false);
      }
      lacc = __builtin_amdgcn_wmma_f32_16x16x32_bf16(
          false, pf, false, __builtin_bit_cast(v16bf, onef),
          (short)0, lacc, false, false);
    }
  }

  // ---- normalize and write out (f32, C-layout; lacc already replicated) ----
#pragma unroll
  for (int r = 0; r < 8; ++r) {
    const float inv = 1.0f / lacc[r];
    const size_t orow = bb + (size_t)(q0 + wv * QW + r + 8 * hb) * D_;
#pragma unroll
    for (int t = 0; t < 4; ++t)
      Op[orow + 16 * t + ln] = oacc[t][r] * inv;
  }
}

extern "C" void kernel_launch(void* const* d_in, const int* in_sizes, int n_in,
                              void* d_out, int out_size, void* d_ws, size_t ws_size,
                              hipStream_t stream) {
  (void)in_sizes; (void)n_in; (void)out_size; (void)d_ws; (void)ws_size;
  const float* Q = (const float*)d_in[0];
  const float* K = (const float*)d_in[1];
  const float* V = (const float*)d_in[2];
  float* O = (float*)d_out;
  dim3 grid(B_ * (L_ / QB));  // 8 * 16 = 128 blocks
  sdpa_flash_wmma_bf16<<<grid, 256, 0, stream>>>(Q, K, V, O);
}
